// MultiLoraConv2d_19361712571086
// MI455X (gfx1250) — matrix operand order repaired and checked by reading, not compiled, not run on hardware
//
#include <hip/hip_runtime.h>

// ---------------------------------------------------------------------------
// Multi-LoRA grouped conv for MI455X (gfx1250, wave32, WMMA bf16 path).
//   Stage 1: delta[t]  = scaling * lora_B[t] @ lora_A[t]          (fp32, tiny)
//   Stage 2: agg[b]    = bf16( W + sum_t alphas[b,t]*delta[t] )   (streaming)
//   Stage 3: out[b]    = conv3x3(x[b], agg[b])  as implicit GEMM
//            M=256 (COUT), N=4096 (H*W), K=2304 (CIN*9), v_wmma_f32_16x16x32_bf16
//
// Roofline: 154.6 GFLOP conv vs ~0.3 GB HBM traffic (~13 us @ 23.3 TB/s);
// bf16 WMMA (16384 FLOP/instr) keeps compute near the memory floor. Per-sample
// x (4.2 MB) and weights (1.2 MB) are L2-resident (192 MB), so im2col re-reads
// stay on-chip.
// ---------------------------------------------------------------------------

typedef __bf16 v16bf __attribute__((ext_vector_type(16)));
typedef __bf16 v8bf  __attribute__((ext_vector_type(8)));
typedef float  v8f   __attribute__((ext_vector_type(8)));

union BF16x16 { v16bf v; v8bf h[2]; };

constexpr int B_   = 32;
constexpr int T_   = 4;
constexpr int CIN  = 256;
constexpr int COUT = 256;
constexpr int KSZ  = 3;
constexpr int H_   = 64;
constexpr int W_   = 64;
constexpr int KDIM = CIN * KSZ * KSZ;   // 2304
constexpr int NSP  = H_ * W_;           // 4096
constexpr int RK   = 24;                // R*K
constexpr int MAT  = COUT * KSZ;        // 768 (== CIN*KSZ)
constexpr float SCALING = 2.0f;         // ALPHA / R

__device__ inline unsigned short f2bf(float f) {
  unsigned u = __float_as_uint(f);
  u += 0x7FFFu + ((u >> 16) & 1u);      // round-to-nearest-even
  return (unsigned short)(u >> 16);
}

// ---------------- Stage 1: delta[t] = SCALING * (B_t @ A_t) -----------------
__global__ __launch_bounds__(256) void delta_kernel(const float* __restrict__ lA,
                                                    const float* __restrict__ lB,
                                                    float* __restrict__ delta) {
  int idx = blockIdx.x * 256 + threadIdx.x;          // T*768*768 threads exactly
  int t   = idx / (MAT * MAT);
  int rem = idx - t * MAT * MAT;
  int r   = rem / MAT;
  int s   = rem - r * MAT;
  const float* bp = lB + ((size_t)t * MAT + r) * RK; // lora_B[t][r][:]
  const float* ap = lA + (size_t)t * RK * MAT + s;   // lora_A[t][:][s]
  float acc = 0.f;
#pragma unroll
  for (int k = 0; k < RK; ++k) acc += bp[k] * ap[(size_t)k * MAT];
  delta[idx] = acc * SCALING;
}

// ------- Stage 2: agg[b] = bf16(W + sum_t alphas[b,t]*delta[t]) -------------
__global__ __launch_bounds__(256) void agg_kernel(const float* __restrict__ Wt,
                                                  const float* __restrict__ alphas,
                                                  const float* __restrict__ delta,
                                                  unsigned short* __restrict__ agg) {
  int idx  = blockIdx.x * 256 + threadIdx.x;         // B*COUT*KDIM threads exactly
  int b    = idx / (COUT * KDIM);
  int widx = idx - b * (COUT * KDIM);
  float v = Wt[widx];
#pragma unroll
  for (int t = 0; t < T_; ++t)
    v += alphas[b * T_ + t] * delta[(size_t)t * COUT * KDIM + widx];
  agg[idx] = f2bf(v);
}

// ---------------- Stage 3: implicit-GEMM conv, bf16 WMMA --------------------
// Block: 256 threads (8 waves). BM=BN=128, BK=32.
// Waves: 2 (M) x 4 (N); each wave computes 64x32 = 4x2 WMMA tiles.
// LDS rows padded to 40 ushorts (80B) -> 16B-aligned b128 fragment loads.
__global__ __launch_bounds__(256) void conv_kernel(const float* __restrict__ x,
                                                   const unsigned short* __restrict__ aggw,
                                                   float* __restrict__ out) {
  __shared__ __align__(16) unsigned short As[128 * 40];  // weights tile [m][k]
  __shared__ __align__(16) unsigned short Bs[128 * 40];  // im2col tile [n][k]

  const int tid  = threadIdx.x;
  const int bid  = blockIdx.x;                 // 32 * 2 * 32 blocks
  const int b    = bid >> 6;
  const int rem  = bid & 63;
  const int row0 = (rem >> 5) * 128;           // COUT tile base
  const int n0   = (rem & 31) * 128;           // spatial tile base
  const int h0   = n0 >> 6;

  const int wave   = tid >> 5, lane = tid & 31;
  const int wave_m = wave >> 2, wave_n = wave & 3;
  const int g      = lane >> 4, m16 = lane & 15;

  const float* xb = x + (size_t)b * CIN * NSP;

  // A-tile global->LDS mapping: 2 threads per row, 2x uint4 (16 bf16) each.
  const int arow = tid >> 1;
  const int aseg = (tid & 1) * 2;
  const unsigned short* wrow =
      aggw + ((size_t)b * COUT + row0 + arow) * KDIM;

  // B-tile mapping: thread handles 2 consecutive k for 8 n-positions.
  const int kk2 = tid >> 4;                    // 0..15 -> local k = 2*kk2, +1
  const int dn0 = tid & 15;

  const v8f vzero = {0.f, 0.f, 0.f, 0.f, 0.f, 0.f, 0.f, 0.f};
  v8f acc[4][2];
#pragma unroll
  for (int i = 0; i < 4; ++i)
#pragma unroll
    for (int j = 0; j < 2; ++j) acc[i][j] = vzero;

  for (int kt = 0; kt < KDIM / 32; ++kt) {
    // ---- global fetch (before barrier: overlaps prior iter's WMMAs) ----
    const uint4* ap = (const uint4*)(wrow + kt * 32);
    uint4 a0 = ap[aseg];
    uint4 a1 = ap[aseg + 1];

    // Prefetch next K-step's A chunk into L2/L0 (global_prefetch_b8).
    if (kt + 1 < KDIM / 32)
      __builtin_prefetch((const void*)(wrow + (kt + 1) * 32), 0, 1);

    int k0  = kt * 32 + kk2 * 2;
    int ci0 = k0 / 9, r0 = k0 - ci0 * 9;
    int kh0 = r0 / 3, kw0 = r0 - kh0 * 3;
    int k1  = k0 + 1;
    int ci1 = k1 / 9, r1 = k1 - ci1 * 9;
    int kh1 = r1 / 3, kw1 = r1 - kh1 * 3;
    const int dh0 = kh0 - 1, dw0 = kw0 - 1;
    const int dh1 = kh1 - 1, dw1 = kw1 - 1;
    const float* xc0 = xb + ci0 * NSP;
    const float* xc1 = xb + ci1 * NSP;

    // Branchless halo handling: clamp address (always-valid load within the
    // 16KB channel plane) + select 0 -> v_med3/v_cndmask, no exec branches.
    float bv0[8], bv1[8];
#pragma unroll
    for (int j = 0; j < 8; ++j) {
      int dn = dn0 + 16 * j;
      int h  = h0 + (dn >> 6);
      int w  = dn & 63;

      int ih = h + dh0, iw = w + dw0;
      bool ok0 = ((unsigned)ih < (unsigned)H_) & ((unsigned)iw < (unsigned)W_);
      int ihc = ih < 0 ? 0 : (ih > H_ - 1 ? H_ - 1 : ih);
      int iwc = iw < 0 ? 0 : (iw > W_ - 1 ? W_ - 1 : iw);
      float v0 = xc0[ihc * W_ + iwc];
      bv0[j] = ok0 ? v0 : 0.f;

      ih = h + dh1; iw = w + dw1;
      bool ok1 = ((unsigned)ih < (unsigned)H_) & ((unsigned)iw < (unsigned)W_);
      ihc = ih < 0 ? 0 : (ih > H_ - 1 ? H_ - 1 : ih);
      iwc = iw < 0 ? 0 : (iw > W_ - 1 ? W_ - 1 : iw);
      float v1 = xc1[ihc * W_ + iwc];
      bv1[j] = ok1 ? v1 : 0.f;
    }

    __syncthreads();   // previous iteration's fragment reads done

    uint4* aq = (uint4*)&As[arow * 40];
    aq[aseg]     = a0;
    aq[aseg + 1] = a1;

#pragma unroll
    for (int j = 0; j < 8; ++j) {
      int dn = dn0 + 16 * j;
      unsigned pk = (unsigned)f2bf(bv0[j]) | ((unsigned)f2bf(bv1[j]) << 16);
      *(unsigned*)&Bs[dn * 40 + kk2 * 2] = pk;
    }

    __syncthreads();   // tiles visible

    // ---- fragments from LDS (aligned b128 loads) ----
    // A 16x32 bf16: lane group g holds K {8g..8g+7} (V0-3) and {16+8g..23+8g} (V4-7)
    BF16x16 afr[4];
#pragma unroll
    for (int i = 0; i < 4; ++i) {
      int row = (wave_m * 64 + i * 16 + m16) * 40;
      afr[i].h[0] = *(const v8bf*)&As[row + g * 8];
      afr[i].h[1] = *(const v8bf*)&As[row + 16 + g * 8];
    }
    // B 32x16 bf16: lane group g holds K {8g..8g+7} and {16+8g..23+8g}
    BF16x16 bfr[2];
#pragma unroll
    for (int i = 0; i < 2; ++i) {
      int row = (wave_n * 32 + i * 16 + m16) * 40;
      bfr[i].h[0] = *(const v8bf*)&Bs[row + g * 8];
      bfr[i].h[1] = *(const v8bf*)&Bs[row + 16 + g * 8];
    }

#pragma unroll
    for (int i = 0; i < 4; ++i)
#pragma unroll
      for (int j = 0; j < 2; ++j)
        acc[i][j] = __builtin_amdgcn_wmma_f32_16x16x32_bf16(
            false, afr[i].v, false, bfr[j].v, (short)0, acc[i][j],
            false, false);
  }

  // ---- epilogue: C/D layout -> VGPR r holds M = r + 8*g, N = lane&15 ----
#pragma unroll
  for (int i = 0; i < 4; ++i) {
    int co = row0 + wave_m * 64 + i * 16;
#pragma unroll
    for (int j = 0; j < 2; ++j) {
      int n = n0 + wave_n * 32 + j * 16 + m16;
      float* op = out + ((size_t)b * COUT + co) * NSP + n;
#pragma unroll
      for (int r = 0; r < 8; ++r)
        op[(size_t)(r + g * 8) * NSP] = acc[i][j][r];
    }
  }
}

// ---------------------------------------------------------------------------
extern "C" void kernel_launch(void* const* d_in, const int* in_sizes, int n_in,
                              void* d_out, int out_size, void* d_ws, size_t ws_size,
                              hipStream_t stream) {
  const float* x      = (const float*)d_in[0];   // [32,256,64,64]
  const float* alphas = (const float*)d_in[1];   // [32,4]
  const float* W      = (const float*)d_in[2];   // [256,256,3,3]
  const float* lA     = (const float*)d_in[3];   // [4,24,768]
  const float* lB     = (const float*)d_in[4];   // [4,768,24]
  float* out = (float*)d_out;                    // [32,256,64,64] fp32

  // Workspace: delta fp32 (9.44 MB) + agg bf16 (37.75 MB) ~= 47.2 MB
  float* delta_ws = (float*)d_ws;
  unsigned short* agg_ws =
      (unsigned short*)((char*)d_ws + (size_t)T_ * MAT * MAT * sizeof(float));

  delta_kernel<<<(T_ * MAT * MAT) / 256, 256, 0, stream>>>(lA, lB, delta_ws);
  agg_kernel<<<(B_ * COUT * KDIM) / 256, 256, 0, stream>>>(W, alphas, delta_ws,
                                                           agg_ws);
  conv_kernel<<<B_ * 2 * (NSP / 128), 256, 0, stream>>>(x, agg_ws, out);
  (void)in_sizes; (void)n_in; (void)out_size; (void)ws_size;
}
